// PureAEVComputer_80032420593972
// MI455X (gfx1250) — matrix operand order repaired and checked by reading, not compiled, not run on hardware
//
#include <hip/hip_runtime.h>
#include <hip/hip_bf16.h>

typedef __attribute__((ext_vector_type(16))) _Float16 v16h;
typedef __attribute__((ext_vector_type(8)))  float    v8f;
typedef __attribute__((ext_vector_type(4)))  float    v4f;

#define RCR_F    5.2f
#define PI_RCR   (3.14159265358979323846f / 5.2f)
#define ETA      16.0f
#define NATOMS   512
#define NMOL     16
#define NSHF     16
#define OUTW     1008

// One block = 256 threads = 8 wave32 = 8 atoms of one molecule.
// One wave computes the full 4x16 radial AEV of one atom via 16 chained
// v_wmma_f32_16x16x32_f16 ops (A = species one-hot, B = fc*exp radial core),
// then streams the padded 1008-float row out as aligned b128 NT stores.
__global__ __launch_bounds__(256) void aev_radial_wmma(
    const int*   __restrict__ species,
    const float* __restrict__ coords,
    float*       __restrict__ out)
{
    __shared__ float sx[NATOMS], sy[NATOMS], sz[NATOMS];
    __shared__ int   ss[NATOMS];
    __shared__ float sd[8][32];                       // per-wave staged distances
    __shared__ float sf[8][32];                       // per-wave staged cutoffs
    __shared__ __align__(16) float sres[8][64];       // per-wave result staging

    const int b    = blockIdx.x >> 6;   // molecule  (16)
    const int tile = blockIdx.x & 63;   // 64 tiles of 8 atoms
    const int tid  = threadIdx.x;
    const int wave = tid >> 5;
    const int lane = tid & 31;

    // Stage whole molecule into LDS (coords SoA + species)
    for (int j = tid; j < NATOMS; j += 256) {
        const float* cp = coords + ((size_t)(b * NATOMS + j)) * 3;
        sx[j] = cp[0];
        sy[j] = cp[1];
        sz[j] = cp[2];
        ss[j] = species[b * NATOMS + j];
    }
    __syncthreads();

    const int   i      = tile * 8 + wave;       // this wave's atom
    const float xi     = sx[i], yi = sy[i], zi = sz[i];
    const int   r      = lane & 15;             // radial shift index (col N)
    const float shfr   = 0.9f + 0.26875f * (float)r;
    const int   mrow   = lane & 15;             // species row for A (M)
    const int   laneHi = (lane >= 16) ? 1 : 0;

    v8f acc = {};                               // 16x16 f32 C/D accumulator

    #pragma unroll 1
    for (int c = 0; c < NATOMS / 32; ++c) {
        const int j0 = c * 32;

        // Each lane computes (d, fc) for one neighbor of the chunk.
        {
            const int j  = j0 + lane;
            const float dx = xi - sx[j];
            const float dy = yi - sy[j];
            const float dz = zi - sz[j];
            const float sq = dx * dx + dy * dy + dz * dz;
            const float d  = sqrtf(sq);
            const bool  in = (d > 0.0f) && (d <= RCR_F);
            const float fc = in ? (0.5f * __cosf(PI_RCR * d) + 0.5f) : 0.0f;
            sd[wave][lane] = d;
            sf[wave][lane] = fc;
        }
        // Per-wave LDS slice: wave-internal ordering is enough (DS ops are
        // in-order within a wave); just stop compiler reordering.
        __builtin_amdgcn_wave_barrier();

        // Build WMMA operands in contiguous k order (k = 16*half + e, same for A and B).
        v16h A, Bm;
        #pragma unroll
        for (int e = 0; e < 16; ++e) {
            // B-matrix 32x16: lane<16 -> col N=lane, K = e (0..15);
            //                 lane>=16 -> col N=lane-16, K = 16+e.
            const int   kB = (laneHi ? 16 : 0) + e;
            const float d  = sd[wave][kB];
            const float fc = sf[wave][kB];
            const float t  = d - shfr;
            Bm[e] = (_Float16)(fc * __expf(-ETA * t * t));

            // A-matrix 16x32: lane<16 -> row M=lane, K = e (0..15);
            //                 lane>=16 -> row M=lane-16, K = 16+e.
            const int kA = (laneHi ? 16 : 0) + e;
            const int sj = ss[j0 + kA];
            A[e] = (_Float16)(((mrow < 4) && (sj == mrow)) ? 1.0f : 0.0f);
        }

        // D = A(one-hot) x B(radial core) + C   — EXEC all-1s here.
        acc = __builtin_amdgcn_wmma_f32_16x16x32_f16(
                  /*neg_a=*/false, A, /*neg_b=*/false, Bm,
                  /*c_mod=*/(short)0, acc, /*reuse_a=*/false, /*reuse_b=*/false);

        __builtin_amdgcn_wave_barrier();
    }

    // D layout: element p, lanes 0-15 -> row M=p (species), col N=lane (shift).
    // Stage the 64 valid results in LDS so the global write can be pure b128.
    if (lane < 16) {
        #pragma unroll
        for (int p = 0; p < 4; ++p) {
            sres[wave][p * NSHF + r] = acc[p];
        }
    }
    __builtin_amdgcn_wave_barrier();

    // Stream the whole 1008-float row (results + zero pad) as aligned
    // nontemporal b128 stores: 252 float4 per atom, 32 lanes.
    const size_t base = ((size_t)(b * NATOMS + i)) * OUTW;
    v4f* __restrict__ o4 = (v4f*)(out + base);      // 4032B row -> 16B aligned
    const v4f z4 = {0.0f, 0.0f, 0.0f, 0.0f};

    // Peeled first group: the only one that carries results from LDS.
    {
        v4f v = z4;
        if (lane < 16) {
            v = *(const v4f*)&sres[wave][lane * 4]; // ds_load_b128
        }
        __builtin_nontemporal_store(v, o4 + lane);  // global_store_b128 (NT)
    }
    // Pure zero-fill stream for the padded tail.
    #pragma unroll 4
    for (int t = lane + 32; t < OUTW / 4; t += 32) {
        __builtin_nontemporal_store(z4, o4 + t);    // global_store_b128 (NT)
    }
}

extern "C" void kernel_launch(void* const* d_in, const int* in_sizes, int n_in,
                              void* d_out, int out_size, void* d_ws, size_t ws_size,
                              hipStream_t stream) {
    (void)in_sizes; (void)n_in; (void)d_ws; (void)ws_size; (void)out_size;
    const int*   species = (const int*)d_in[0];     // (16, 512) int32
    const float* coords  = (const float*)d_in[1];   // (16, 512, 3) f32
    float*       out     = (float*)d_out;           // (16, 512, 1008) f32

    dim3 grid(NMOL * (NATOMS / 8));   // 1024 blocks
    dim3 block(256);                  // 8 wave32 per block
    hipLaunchKernelGGL(aev_radial_wmma, grid, block, 0, stream,
                       species, coords, out);
}
